// SparseMoEBlock_24189255811639
// MI455X (gfx1250) — compile-verified, matrix operand order checked by reading, not run on hardware
//
#include <hip/hip_runtime.h>
#include <hip/hip_bf16.h>
#include <math.h>

#define T_TOK 4096
#define H_DIM 1024
#define I_DIM 4096
#define E_NUM 8
#define TM 128
#define TN 128
#define TK 32
#define LDSS 40   // LDS row stride in halves (32 data + 8 pad) -> conflict-free b128 reads

typedef __attribute__((ext_vector_type(16))) __bf16        bf16x16;
typedef __attribute__((ext_vector_type(4)))  __bf16        bf16x4;
typedef __attribute__((ext_vector_type(8)))  float         f32x8;
typedef __attribute__((ext_vector_type(4)))  float         f32x4;
typedef __attribute__((ext_vector_type(4)))  unsigned int  u32x4;
typedef __attribute__((ext_vector_type(8)))  unsigned int  u32x8;

#if defined(__has_builtin)
#if __has_builtin(__builtin_amdgcn_global_load_async_to_lds_b128)
#define HAVE_ASYNC_LDS 1
#endif
#endif
#ifndef HAVE_ASYNC_LDS
#define HAVE_ASYNC_LDS 0
#endif

// builtin signature (from hipcc diagnostic): (int4 addrspace(1)*, int4 addrspace(3)*, Imm, Imm)
typedef int v4i __attribute__((vector_size(16)));
typedef v4i __attribute__((address_space(1))) as1_v4i;
typedef v4i __attribute__((address_space(3))) as3_v4i;

// 16B global -> LDS copy. Async path: per-lane DMA tracked by ASYNCcnt (no VGPR staging).
// LDS offset = low 32 bits of the generic shared-aperture address (ISA 10.2).
__device__ __forceinline__ void copy16B_g2lds(unsigned short* lds_dst, const unsigned short* gsrc) {
#if HAVE_ASYNC_LDS
    __builtin_amdgcn_global_load_async_to_lds_b128(
        (as1_v4i*)(size_t)gsrc, (as3_v4i*)(unsigned)(size_t)lds_dst, 0, 0);
#else
    *(u32x4*)lds_dst = *(const u32x4*)gsrc;
#endif
}

__device__ __forceinline__ void wait_async_lds() {
#if HAVE_ASYNC_LDS
    asm volatile("s_wait_asynccnt 0x0" ::: "memory");
#endif
}

// scalar f32 -> bf16 (RNE) for epilogue stores
__device__ __forceinline__ unsigned short f2bf(float f) {
    unsigned u = __builtin_bit_cast(unsigned, f);
    u += 0x7FFFu + ((u >> 16) & 1u);
    return (unsigned short)(u >> 16);
}

// 16 fp32 -> 16 bf16 into LDS via packed v_cvt_pk_bf16_f32
__device__ __forceinline__ void cvt_store16(unsigned short* d, const float* s) {
    #pragma unroll
    for (int j = 0; j < 4; ++j) {
        f32x4 v = *(const f32x4*)(s + j * 4);
        *(bf16x4*)(d + j * 4) = __builtin_convertvector(v, bf16x4);
    }
}

// 16-bit A/B fragment per CDNA5 ISA 7.12.2:
// lanes 0-15: row = lane,    K = {0..7, 16..23}
// lanes16-31: row = lane-16, K = {8..15, 24..31}
__device__ __forceinline__ bf16x16 load_frag(const unsigned short* tile, int lane) {
    const unsigned short* p = tile + (lane & 15) * LDSS + ((lane >> 4) << 3);
    u32x4 lo = *(const u32x4*)(p);
    u32x4 hi = *(const u32x4*)(p + 16);
    u32x8 t;
    t[0]=lo[0]; t[1]=lo[1]; t[2]=lo[2]; t[3]=lo[3];
    t[4]=hi[0]; t[5]=hi[1]; t[6]=hi[2]; t[7]=hi[3];
    return __builtin_bit_cast(bf16x16, t);
}

// ---------------- kernel 0: pre-convert x to bf16 ----------------
__global__ __launch_bounds__(256)
void cvt_x_kernel(const float* __restrict__ x, unsigned short* __restrict__ xb) {
    size_t i = ((size_t)blockIdx.x * 256 + threadIdx.x) * 8;
    f32x4 a = *(const f32x4*)(x + i);
    f32x4 b = *(const f32x4*)(x + i + 4);
    *(bf16x4*)(xb + i)     = __builtin_convertvector(a, bf16x4);
    *(bf16x4*)(xb + i + 4) = __builtin_convertvector(b, bf16x4);
}

// ---------------- kernel 1: router (softmax + top-2 + renorm) ----------------
__global__ __launch_bounds__(256)
void router_kernel(const float* __restrict__ x, const float* __restrict__ gw,
                   int* __restrict__ topk_idx, float* __restrict__ topk_w) {
    int wave = threadIdx.x >> 5, lane = threadIdx.x & 31;
    int token = blockIdx.x * 8 + wave;
    float acc[E_NUM];
    #pragma unroll
    for (int e = 0; e < E_NUM; ++e) acc[e] = 0.f;
    const float* xr = x + (size_t)token * H_DIM;
    #pragma unroll
    for (int i = 0; i < H_DIM / 128; ++i) {
        int h = i * 128 + lane * 4;
        float4 xv = *(const float4*)(xr + h);
        #pragma unroll
        for (int e = 0; e < E_NUM; ++e) {
            float4 gv = *(const float4*)(gw + e * H_DIM + h);
            acc[e] += xv.x*gv.x + xv.y*gv.y + xv.z*gv.z + xv.w*gv.w;
        }
    }
    #pragma unroll
    for (int e = 0; e < E_NUM; ++e)
        #pragma unroll
        for (int off = 16; off > 0; off >>= 1)
            acc[e] += __shfl_xor(acc[e], off, 32);
    if (lane == 0) {
        float m = acc[0];
        #pragma unroll
        for (int e = 1; e < E_NUM; ++e) m = fmaxf(m, acc[e]);
        float p[E_NUM], s = 0.f;
        #pragma unroll
        for (int e = 0; e < E_NUM; ++e) { p[e] = expf(acc[e] - m); s += p[e]; }
        float inv_s = 1.f / s;
        #pragma unroll
        for (int e = 0; e < E_NUM; ++e) p[e] *= inv_s;
        int i0 = 0;
        #pragma unroll
        for (int e = 1; e < E_NUM; ++e) if (p[e] > p[i0]) i0 = e;
        int i1 = (i0 == 0) ? 1 : 0;
        #pragma unroll
        for (int e = 0; e < E_NUM; ++e) if (e != i0 && p[e] > p[i1]) i1 = e;
        float w0 = p[i0], w1 = p[i1], inv = 1.f / (w0 + w1);
        topk_idx[token*2]   = i0;  topk_idx[token*2+1] = i1;
        topk_w[token*2]     = w0*inv; topk_w[token*2+1] = w1*inv;
    }
}

// ---------------- kernel 2: deterministic per-expert compaction ----------------
__global__ __launch_bounds__(256)
void compact_kernel(const int* __restrict__ topk_idx, const float* __restrict__ topk_w,
                    int* __restrict__ cnt_g, int* __restrict__ base_g,
                    int* __restrict__ row_token, int* __restrict__ row_kk,
                    float* __restrict__ row_w) {
    __shared__ int s_cnt[E_NUM], s_base[E_NUM];
    int e = threadIdx.x >> 5, lane = threadIdx.x & 31;
    unsigned lt = (lane == 0) ? 0u : (0xFFFFFFFFu >> (32 - lane));
    int c = 0;
    for (int t0 = 0; t0 < T_TOK; t0 += 32) {
        int t = t0 + lane;
        bool sel = (topk_idx[t*2] == e) || (topk_idx[t*2+1] == e);
        c += __popc((unsigned)__ballot(sel));
    }
    if (lane == 0) s_cnt[e] = c;
    __syncthreads();
    if (threadIdx.x == 0) {
        int run = 0;
        for (int i = 0; i < E_NUM; ++i) {
            s_base[i] = run; run += s_cnt[i];
            cnt_g[i] = s_cnt[i]; base_g[i] = s_base[i];
        }
    }
    __syncthreads();
    int offs = s_base[e];
    for (int t0 = 0; t0 < T_TOK; t0 += 32) {
        int t = t0 + lane;
        int k = (topk_idx[t*2] == e) ? 0 : ((topk_idx[t*2+1] == e) ? 1 : -1);
        unsigned mask = (unsigned)__ballot(k >= 0);
        if (k >= 0) {
            int pos = offs + __popc(mask & lt);
            row_token[pos] = t; row_kk[pos] = k; row_w[pos] = topk_w[t*2+k];
        }
        offs += __popc(mask);
    }
}

// -------- kernel 3: grouped GEMM gate_up (bf16 WMMA) + fused SwiGLU --> act --------
__global__ __launch_bounds__(256)
void gemm_gateup_kernel(const unsigned short* __restrict__ xb, const float* __restrict__ gup,
                        const int* __restrict__ cnt_g, const int* __restrict__ base_g,
                        const int* __restrict__ row_token,
                        unsigned short* __restrict__ act) {
    int e = blockIdx.z;
    int cnt = cnt_g[e];
    int mTile = blockIdx.y;
    if (mTile * TM >= cnt) return;
    int base = base_g[e];
    int nBlock = blockIdx.x * TN;     // over I_DIM

    __shared__ unsigned short sA[TM * LDSS];
    __shared__ unsigned short sG[TN * LDSS];
    __shared__ unsigned short sU[TN * LDSS];

    int tid = threadIdx.x, lane = tid & 31, wave = tid >> 5;
    int waveM = wave & 1, waveN = wave >> 1;

    int lr = tid >> 1;            // loader row 0..127
    int lh = (tid & 1) * 16;      // loader col offset {0,16}
    int grow = mTile * TM + lr;
    int cr = (grow < cnt) ? grow : cnt - 1;   // clamp; out-of-range D rows are discarded
    const unsigned short* aRow = xb + (size_t)row_token[base + cr] * H_DIM;
    const float* gSrc = gup + ((size_t)e * 2 * I_DIM + (nBlock + lr)) * H_DIM;
    const float* uSrc = gup + ((size_t)e * 2 * I_DIM + (I_DIM + nBlock + lr)) * H_DIM;

    f32x8 zero = {0.f,0.f,0.f,0.f,0.f,0.f,0.f,0.f};
    f32x8 accG[4][2], accU[4][2];
    #pragma unroll
    for (int i = 0; i < 4; ++i)
        #pragma unroll
        for (int j = 0; j < 2; ++j) { accG[i][j] = zero; accU[i][j] = zero; }

    for (int kk = 0; kk < H_DIM; kk += TK) {
        __syncthreads();
        if (kk + TK < H_DIM) {   // global_prefetch_b8 of next weight K-tile
            __builtin_prefetch(gSrc + kk + TK + lh, 0, 0);
            __builtin_prefetch(uSrc + kk + TK + lh, 0, 0);
        }
        // A: gathered bf16 rows, async global->LDS DMA (ASYNCcnt)
        copy16B_g2lds(&sA[lr * LDSS + lh],     aRow + kk + lh);
        copy16B_g2lds(&sA[lr * LDSS + lh + 8], aRow + kk + lh + 8);
        // B: fp32 weights -> packed bf16 cvt -> LDS
        cvt_store16(&sG[lr * LDSS + lh], gSrc + kk + lh);
        cvt_store16(&sU[lr * LDSS + lh], uSrc + kk + lh);
        wait_async_lds();
        __syncthreads();

        bf16x16 aF[4], gF[2], uF[2];
        #pragma unroll
        for (int mi = 0; mi < 4; ++mi)
            aF[mi] = load_frag(&sA[(waveM * 64 + mi * 16) * LDSS], lane);
        #pragma unroll
        for (int ni = 0; ni < 2; ++ni) {
            gF[ni] = load_frag(&sG[(waveN * 32 + ni * 16) * LDSS], lane);
            uF[ni] = load_frag(&sU[(waveN * 32 + ni * 16) * LDSS], lane);
        }
        #pragma unroll
        for (int mi = 0; mi < 4; ++mi)
            #pragma unroll
            for (int ni = 0; ni < 2; ++ni) {
                accG[mi][ni] = __builtin_amdgcn_wmma_f32_16x16x32_bf16(
                    false, aF[mi], false, gF[ni], (short)0, accG[mi][ni], false, false);
                accU[mi][ni] = __builtin_amdgcn_wmma_f32_16x16x32_bf16(
                    false, aF[mi], false, uF[ni], (short)0, accU[mi][ni], false, false);
            }
    }

    // D layout: vgpr r -> M = r + 8*(lane>=16), N = lane&15
    int mLane = (lane >> 4) * 8, nLane = lane & 15;
    #pragma unroll
    for (int mi = 0; mi < 4; ++mi)
        #pragma unroll
        for (int r = 0; r < 8; ++r) {
            int m = waveM * 64 + mi * 16 + mLane + r;
            int g2 = mTile * TM + m;
            if (g2 < cnt) {
                #pragma unroll
                for (int ni = 0; ni < 2; ++ni) {
                    int n = waveN * 32 + ni * 16 + nLane;
                    float g = accG[mi][ni][r], u = accU[mi][ni][r];
                    float a = (g / (1.f + expf(-g))) * u;   // silu(g)*u
                    act[(size_t)(base + g2) * I_DIM + (nBlock + n)] = f2bf(a);
                }
            }
        }
}

// -------- kernel 4: grouped GEMM down-proj (bf16 WMMA), weighted scatter --------
__global__ __launch_bounds__(256)
void gemm_down_kernel(const unsigned short* __restrict__ act, const float* __restrict__ dn,
                      const int* __restrict__ cnt_g, const int* __restrict__ base_g,
                      const int* __restrict__ row_token, const int* __restrict__ row_kk,
                      const float* __restrict__ row_w, float* __restrict__ partial) {
    int e = blockIdx.z;
    int cnt = cnt_g[e];
    int mTile = blockIdx.y;
    if (mTile * TM >= cnt) return;
    int base = base_g[e];
    int hBlock = blockIdx.x * TN;     // over H_DIM

    __shared__ unsigned short sA[TM * LDSS];
    __shared__ unsigned short sB[TN * LDSS];

    int tid = threadIdx.x, lane = tid & 31, wave = tid >> 5;
    int waveM = wave & 1, waveN = wave >> 1;

    int lr = tid >> 1, lh = (tid & 1) * 16;
    int grow = mTile * TM + lr;
    int cr = (grow < cnt) ? grow : cnt - 1;
    const unsigned short* aSrc = act + (size_t)(base + cr) * I_DIM;
    const float* bSrc = dn + ((size_t)e * H_DIM + (hBlock + lr)) * I_DIM;

    f32x8 zero = {0.f,0.f,0.f,0.f,0.f,0.f,0.f,0.f};
    f32x8 acc[4][2];
    #pragma unroll
    for (int i = 0; i < 4; ++i)
        #pragma unroll
        for (int j = 0; j < 2; ++j) acc[i][j] = zero;

    for (int kk = 0; kk < I_DIM; kk += TK) {
        __syncthreads();
        if (kk + TK < I_DIM) __builtin_prefetch(bSrc + kk + TK + lh, 0, 0);
        // A: activation tile is already bf16 -> async global->LDS DMA
        copy16B_g2lds(&sA[lr * LDSS + lh],     aSrc + kk + lh);
        copy16B_g2lds(&sA[lr * LDSS + lh + 8], aSrc + kk + lh + 8);
        // B: fp32 down-proj -> packed bf16 cvt -> LDS
        cvt_store16(&sB[lr * LDSS + lh], bSrc + kk + lh);
        wait_async_lds();
        __syncthreads();

        bf16x16 aF[4], bF[2];
        #pragma unroll
        for (int mi = 0; mi < 4; ++mi)
            aF[mi] = load_frag(&sA[(waveM * 64 + mi * 16) * LDSS], lane);
        #pragma unroll
        for (int ni = 0; ni < 2; ++ni)
            bF[ni] = load_frag(&sB[(waveN * 32 + ni * 16) * LDSS], lane);
        #pragma unroll
        for (int mi = 0; mi < 4; ++mi)
            #pragma unroll
            for (int ni = 0; ni < 2; ++ni)
                acc[mi][ni] = __builtin_amdgcn_wmma_f32_16x16x32_bf16(
                    false, aF[mi], false, bF[ni], (short)0, acc[mi][ni], false, false);
    }

    int mLane = (lane >> 4) * 8, nLane = lane & 15;
    #pragma unroll
    for (int mi = 0; mi < 4; ++mi)
        #pragma unroll
        for (int r = 0; r < 8; ++r) {
            int m = waveM * 64 + mi * 16 + mLane + r;
            int g2 = mTile * TM + m;
            if (g2 < cnt) {
                int ridx = base + g2;
                int token = row_token[ridx];
                int ksel  = row_kk[ridx];
                float w   = row_w[ridx];
                size_t dst = (size_t)ksel * T_TOK * H_DIM + (size_t)token * H_DIM + hBlock;
                #pragma unroll
                for (int ni = 0; ni < 2; ++ni) {
                    int n = waveN * 32 + ni * 16 + nLane;
                    partial[dst + n] = w * acc[mi][ni][r];
                }
            }
        }
}

// ---------------- kernel 5: combine two partial buffers ----------------
__global__ __launch_bounds__(256)
void combine_kernel(const float* __restrict__ partial, float* __restrict__ out) {
    size_t i = ((size_t)blockIdx.x * 256 + threadIdx.x) * 4;
    float4 a = *(const float4*)(partial + i);
    float4 b = *(const float4*)(partial + (size_t)T_TOK * H_DIM + i);
    float4 r; r.x = a.x + b.x; r.y = a.y + b.y; r.z = a.z + b.z; r.w = a.w + b.w;
    *(float4*)(out + i) = r;
}

extern "C" void kernel_launch(void* const* d_in, const int* in_sizes, int n_in,
                              void* d_out, int out_size, void* d_ws, size_t ws_size,
                              hipStream_t stream) {
    const float* x   = (const float*)d_in[0];   // (T, H)
    const float* gw  = (const float*)d_in[1];   // (E, H)
    const float* gup = (const float*)d_in[2];   // (E, 2I, H)
    const float* dn  = (const float*)d_in[3];   // (E, H, I)
    float* out = (float*)d_out;

    char* w = (char*)d_ws;
    size_t off = 0;
    auto alloc = [&](size_t bytes) { void* p = w + off; off = (off + bytes + 255) & ~(size_t)255; return p; };
    int*            topk_idx  = (int*)           alloc((size_t)T_TOK * 2 * 4);
    float*          topk_wt   = (float*)         alloc((size_t)T_TOK * 2 * 4);
    int*            cnt_g     = (int*)           alloc(E_NUM * 4);
    int*            base_g    = (int*)           alloc(E_NUM * 4);
    int*            row_token = (int*)           alloc((size_t)T_TOK * 2 * 4);
    int*            row_kk    = (int*)           alloc((size_t)T_TOK * 2 * 4);
    float*          row_w     = (float*)         alloc((size_t)T_TOK * 2 * 4);
    unsigned short* xb        = (unsigned short*)alloc((size_t)T_TOK * H_DIM * 2);
    unsigned short* act       = (unsigned short*)alloc((size_t)T_TOK * 2 * I_DIM * 2);
    float*          partial   = (float*)         alloc((size_t)2 * T_TOK * H_DIM * 4);

    cvt_x_kernel<<<(T_TOK * H_DIM / 8) / 256, 256, 0, stream>>>(x, xb);
    router_kernel<<<T_TOK / 8, 256, 0, stream>>>(x, gw, topk_idx, topk_wt);
    compact_kernel<<<1, 256, 0, stream>>>(topk_idx, topk_wt, cnt_g, base_g,
                                          row_token, row_kk, row_w);
    gemm_gateup_kernel<<<dim3(I_DIM / TN, T_TOK / TM, E_NUM), 256, 0, stream>>>(
        xb, gup, cnt_g, base_g, row_token, act);
    gemm_down_kernel<<<dim3(H_DIM / TN, T_TOK / TM, E_NUM), 256, 0, stream>>>(
        act, dn, cnt_g, base_g, row_token, row_kk, row_w, partial);
    combine_kernel<<<(T_TOK * H_DIM / 4) / 256, 256, 0, stream>>>(partial, out);
}
